// AWDLSTM_55516747268194
// MI455X (gfx1250) — matrix operand order repaired and checked by reading, not compiled, not run on hardware
//
#include <hip/hip_runtime.h>

// ---------------------------------------------------------------------------
// AWD-LSTM forward for MI455X (gfx1250, wave32, WMMA bf16)
// V=33278 E=400 H=1150 T=70 B=80
// ---------------------------------------------------------------------------

typedef __attribute__((ext_vector_type(16))) __bf16 v16bf;
typedef __attribute__((ext_vector_type(8)))  __bf16 v8bf;
typedef __attribute__((ext_vector_type(8)))  float  v8f;

#define TT 70
#define BB 80
#define VV 33278
#define EE 400

__device__ __forceinline__ unsigned short f32_to_bf16_bits(float f) {
    unsigned int u = __float_as_uint(f);
    unsigned int r = u + 0x7FFFu + ((u >> 16) & 1u);   // round-to-nearest-even
    return (unsigned short)(r >> 16);
}

__device__ __forceinline__ float sigmoid_f(float x) {
    return 1.0f / (1.0f + __expf(-x));
}
__device__ __forceinline__ float tanh_f(float x) {
    x = fminf(15.0f, fmaxf(-15.0f, x));
    float e = __expf(2.0f * x);
    return (e - 1.0f) / (e + 1.0f);
}

__device__ __forceinline__ v8f wmma_bf16(v16bf a, v16bf b, v8f c) {
    return __builtin_amdgcn_wmma_f32_16x16x32_bf16(false, a, false, b,
                                                   (short)0, c, false, false);
}

// A-matrix 16x32 bf16 tile load (ISA 7.12.2): lane half selects K runs
// {half*8 .. half*8+8} and {16+half*8 .. 16+half*8+8}; two 16B loads.
__device__ __forceinline__ v16bf load_A_tile(const unsigned short* __restrict__ A,
                                             int lda, int rowBase, int ktile, int lane) {
    int m = lane & 15, half = lane >> 4;
    const unsigned short* p = A + (size_t)(rowBase + m) * lda + ktile * 32 + half * 8;
    union { v16bf v; v8bf h[2]; } u;
    u.h[0] = *reinterpret_cast<const v8bf*>(p);
    u.h[1] = *reinterpret_cast<const v8bf*>(p + 16);
    return u.v;
}

// B-matrix 32x16 bf16 tile: lane holds column n=lane&15, K runs
// {half*8 .. half*8+8} and {16+half*8 .. 16+half*8+8} (same as A tile).
// B[k][n] = W[n][k] with W row-major [Nrows][K] -> two 16B loads per lane.
__device__ __forceinline__ v16bf load_B_tile(const unsigned short* __restrict__ W,
                                             int ldb, int nBase, int ktile, int lane) {
    int n = lane & 15, half = lane >> 4;
    const unsigned short* p = W + (size_t)(nBase + n) * ldb + ktile * 32 + half * 8;
    union { v16bf v; v8bf h[2]; } u;
    u.h[0] = *reinterpret_cast<const v8bf*>(p);
    u.h[1] = *reinterpret_cast<const v8bf*>(p + 16);
    return u.v;
}

// fragment set for one 32x64x32 k-step (2 A tiles, 4 B tiles)
struct Frag { v16bf a0, a1, b[4]; };

__device__ __forceinline__ void load_frag(Frag& f,
        const unsigned short* __restrict__ A, int lda, int mBase,
        const unsigned short* __restrict__ Bw, int ldb, int nBase,
        int kt, int lane) {
    f.a0 = load_A_tile(A, lda, mBase,      kt, lane);
    f.a1 = load_A_tile(A, lda, mBase + 16, kt, lane);
    #pragma unroll
    for (int j = 0; j < 4; ++j)
        f.b[j] = load_B_tile(Bw, ldb, nBase + 16 * j, kt, lane);
}

__device__ __forceinline__ void frag_mma(v8f (&acc)[2][4], const Frag& f) {
    #pragma unroll
    for (int j = 0; j < 4; ++j) {
        acc[0][j] = wmma_bf16(f.a0, f.b[j], acc[0][j]);
        acc[1][j] = wmma_bf16(f.a1, f.b[j], acc[1][j]);
    }
}

// ---------------------------------------------------------------------------
// Generic f32->bf16 convert with zero padding: src[R][C] -> dst[Rp][Cp]
// ---------------------------------------------------------------------------
__global__ void convert_pad_bf16_kernel(const float* __restrict__ src, int R, int C,
                                        unsigned short* __restrict__ dst, int Rp, int Cp) {
    long total = (long)Rp * Cp;
    long step  = (long)gridDim.x * blockDim.x;
    for (long i = (long)blockIdx.x * blockDim.x + threadIdx.x; i < total; i += step) {
        int r = (int)(i / Cp), c = (int)(i % Cp);
        float v = (r < R && c < C) ? src[(size_t)r * C + c] : 0.0f;
        dst[i] = f32_to_bf16_bits(v);
    }
}

__global__ void zero_u16_kernel(unsigned short* __restrict__ p, long n) {
    long step = (long)gridDim.x * blockDim.x;
    for (long i = (long)blockIdx.x * blockDim.x + threadIdx.x; i < n; i += step) p[i] = 0;
}

// ---------------------------------------------------------------------------
// Embedding gather -> bf16 padded [T*B][Cp]
// ---------------------------------------------------------------------------
__global__ void embed_kernel(const int* __restrict__ tokens, const float* __restrict__ embW,
                             unsigned short* __restrict__ dst, int Cp) {
    int r = blockIdx.x;                       // 0..5599
    int tok = tokens[r];
    const float* src = embW + (size_t)tok * EE;
    for (int c = threadIdx.x; c < Cp; c += blockDim.x)
        dst[(size_t)r * Cp + c] = f32_to_bf16_bits(c < EE ? src[c] : 0.0f);
}

// ---------------------------------------------------------------------------
// init hidden/cell state for a layer
// ---------------------------------------------------------------------------
__global__ void init_state_kernel(const float* __restrict__ h0, const float* __restrict__ c0,
                                  int H, int Kp, unsigned short* __restrict__ h_bf,
                                  float* __restrict__ h_state, float* __restrict__ c_state) {
    int i = blockIdx.x * blockDim.x + threadIdx.x;
    if (i >= BB * H) return;
    int b = i / H, c = i % H;
    float hv = h0[i];
    h_bf[(size_t)b * Kp + c] = f32_to_bf16_bits(hv);
    h_state[i] = hv;
    c_state[i] = c0[i];
}

// ---------------------------------------------------------------------------
// WMMA GEMM: C[M][N] = A_bf[M][Kp] * Bw_bf[N][Kp]^T + bias0 + bias1
// one wave per block, 32x64 output per wave (2 M-tiles x 4 N-tiles: 8 WMMAs
// per k-step on 6 tile-loads). Ping-pong fragment registers, manually
// unrolled by 2: loads for k+1 issue before the 8 WMMAs of k, and each
// fragment is consumed from the registers it was loaded into (no v_mov copies).
// nt=1 -> non-temporal stores (decoder: 745MB streaming output, keep L2 for weights)
// ---------------------------------------------------------------------------
__global__ __launch_bounds__(32)
void wmma_gemm_bias_kernel(const unsigned short* __restrict__ A, int lda,
                           const unsigned short* __restrict__ Bw, int ldb,
                           const float* __restrict__ bias0, const float* __restrict__ bias1,
                           float* __restrict__ C, long ldc, int N, int ktiles, int nt) {
    int lane  = threadIdx.x;
    int mBase = blockIdx.x * 32;
    int nBase = blockIdx.y * 64;

    v8f acc[2][4];
    #pragma unroll
    for (int i = 0; i < 2; ++i)
        #pragma unroll
        for (int j = 0; j < 4; ++j) acc[i][j] = (v8f)(0.0f);

    Frag f0, f1;
    load_frag(f0, A, lda, mBase, Bw, ldb, nBase, 0, lane);   // ktiles >= 2 always

    int kt = 0;
    for (; kt + 2 < ktiles; kt += 2) {
        load_frag(f1, A, lda, mBase, Bw, ldb, nBase, kt + 1, lane);
        frag_mma(acc, f0);
        load_frag(f0, A, lda, mBase, Bw, ldb, nBase, kt + 2, lane);
        frag_mma(acc, f1);
    }
    if (kt + 1 < ktiles) {            // two tiles remain
        load_frag(f1, A, lda, mBase, Bw, ldb, nBase, kt + 1, lane);
        frag_mma(acc, f0);
        frag_mma(acc, f1);
    } else {                          // one tile remains
        frag_mma(acc, f0);
    }

    int nlane = lane & 15, half = lane >> 4;
    #pragma unroll
    for (int j = 0; j < 4; ++j) {
        int n = nBase + 16 * j + nlane;
        if (n >= N) continue;
        float bsum = 0.0f;
        if (bias0) bsum += bias0[n];
        if (bias1) bsum += bias1[n];
        #pragma unroll
        for (int i = 0; i < 2; ++i) {
            #pragma unroll
            for (int v = 0; v < 8; ++v) {
                int m = mBase + i * 16 + half * 8 + v;
                float val = acc[i][j][v] + bsum;
                float* dst = C + (size_t)m * ldc + n;
                if (nt) __builtin_nontemporal_store(val, dst);
                else    *dst = val;
            }
        }
    }
}

// ---------------------------------------------------------------------------
// Fused LSTM step: gates[B][4H] = xW_t + h_in * Whh^T, then gate math.
// One workgroup per 16-column chunk of H; 4 waves = gates i,f,g,o.
// Each wave: 5 M-tiles (B=80) x 1 N-tile, K-loop of WMMA bf16.
// Gates staged in LDS, then elementwise c/h update; h written f32 + bf16
// (ping-pong feedback buffer + next layer's activation buffer).
// ---------------------------------------------------------------------------
__global__ __launch_bounds__(128)
void lstm_step_kernel(const float* __restrict__ xW, int t,
                      const unsigned short* __restrict__ Whh,
                      const unsigned short* __restrict__ h_in,
                      int Kp, int H, int ktiles,
                      unsigned short* __restrict__ h_out,
                      float* __restrict__ c_state, float* __restrict__ h_state,
                      unsigned short* __restrict__ x_out) {
    __shared__ float lds[4 * BB * 16];          // 20 KB

    int lane = threadIdx.x & 31;
    int gate = threadIdx.x >> 5;                // wave id = gate index (i,f,g,o)
    int col0 = blockIdx.x * 16;
    int nlane = lane & 15, half = lane >> 4;

    // init accumulators from the precomputed input contribution (xW + biases)
    long xw_row0 = (long)t * BB * 4608;
    int  colG  = gate * H + col0 + nlane;
    bool colOK = (col0 + nlane) < H;

    v8f acc[5];
    #pragma unroll
    for (int mt = 0; mt < 5; ++mt) {
        #pragma unroll
        for (int v = 0; v < 8; ++v) {
            int b = mt * 16 + half * 8 + v;
            acc[mt][v] = colOK ? xW[xw_row0 + (long)b * 4608 + colG] : 0.0f;
        }
    }

    int wrow = gate * H + col0;                 // Whh base row for this gate chunk
    for (int kt = 0; kt < ktiles; ++kt) {
        v16bf bmat = load_B_tile(Whh, Kp, wrow, kt, lane);
        #pragma unroll
        for (int mt = 0; mt < 5; ++mt) {
            v16bf amat = load_A_tile(h_in, Kp, mt * 16, kt, lane);
            acc[mt] = wmma_bf16(amat, bmat, acc[mt]);
        }
    }

    // stage gates to LDS: lds[gate][b][j]
    #pragma unroll
    for (int mt = 0; mt < 5; ++mt) {
        #pragma unroll
        for (int v = 0; v < 8; ++v) {
            int b = mt * 16 + half * 8 + v;
            lds[(gate * BB + b) * 16 + nlane] = acc[mt][v];
        }
    }
    __syncthreads();

    // elementwise LSTM update over 80x16 chunk
    long xout_row0 = (long)t * BB * Kp;
    for (int e = threadIdx.x; e < BB * 16; e += 128) {
        int b = e >> 4, j = e & 15;
        int col = col0 + j;
        if (col >= H) continue;
        float iv = lds[(0 * BB + b) * 16 + j];
        float fv = lds[(1 * BB + b) * 16 + j];
        float gv = lds[(2 * BB + b) * 16 + j];
        float ov = lds[(3 * BB + b) * 16 + j];
        long  si = (long)b * H + col;
        float cprev = c_state[si];
        float cnew  = sigmoid_f(fv) * cprev + sigmoid_f(iv) * tanh_f(gv);
        float hnew  = sigmoid_f(ov) * tanh_f(cnew);
        c_state[si] = cnew;
        h_state[si] = hnew;
        unsigned short hb = f32_to_bf16_bits(hnew);
        h_out[(long)b * Kp + col] = hb;
        x_out[xout_row0 + (long)b * Kp + col] = hb;
    }
}

// ---------------------------------------------------------------------------
// host orchestration
// ---------------------------------------------------------------------------
extern "C" void kernel_launch(void* const* d_in, const int* in_sizes, int n_in,
                              void* d_out, int out_size, void* d_ws, size_t ws_size,
                              hipStream_t stream) {
    (void)in_sizes; (void)n_in; (void)out_size; (void)ws_size;

    const int*   tokens = (const int*)  d_in[0];
    const float* embW   = (const float*)d_in[1];
    const float* decW   = (const float*)d_in[2];
    const float* decb   = (const float*)d_in[3];
    const float *Wih[3], *Whh[3], *bih[3], *bhh[3], *h0[3], *c0[3];
    for (int l = 0; l < 3; ++l) {
        Wih[l] = (const float*)d_in[4 + 6 * l + 0];
        Whh[l] = (const float*)d_in[4 + 6 * l + 1];
        bih[l] = (const float*)d_in[4 + 6 * l + 2];
        bhh[l] = (const float*)d_in[4 + 6 * l + 3];
        h0[l]  = (const float*)d_in[4 + 6 * l + 4];
        c0[l]  = (const float*)d_in[4 + 6 * l + 5];
    }

    const int Hs[3]  = {1150, 1150, 400};    // hidden size per layer
    const int Is[3]  = {400, 1150, 1150};    // input size per layer
    const int IKp[3] = {416, 1152, 1152};    // pad32(I)
    const int HKp[3] = {1152, 1152, 416};    // pad32(H)
    const int G4[3]  = {4600, 4600, 1600};   // 4H
    const int Grp[3] = {4608, 4608, 1600};   // ceil(4H/64)*64 : Wih row pad
    const int Wrp[3] = {4664, 4664, 1664};   // 4H+64 : Whh row pad (tile-safe)

    // ---- workspace bump allocator (256B aligned) ----
    char* wsp = (char*)d_ws;
    auto alloc = [&](size_t bytes) -> void* {
        void* p = (void*)wsp;
        wsp += (bytes + 255) & ~(size_t)255;
        return p;
    };
    unsigned short* WihB[3]; unsigned short* WhhB[3];
    for (int l = 0; l < 3; ++l) {
        WihB[l] = (unsigned short*)alloc((size_t)Grp[l] * IKp[l] * 2);
        WhhB[l] = (unsigned short*)alloc((size_t)Wrp[l] * HKp[l] * 2);
    }
    unsigned short* decWB = (unsigned short*)alloc((size_t)33280 * 416 * 2);
    unsigned short* xb[4];
    xb[0] = (unsigned short*)alloc((size_t)TT * BB * 416 * 2);
    xb[1] = (unsigned short*)alloc((size_t)TT * BB * 1152 * 2);
    xb[2] = (unsigned short*)alloc((size_t)TT * BB * 1152 * 2);
    xb[3] = (unsigned short*)alloc((size_t)TT * BB * 416 * 2);
    float* xWbuf = (float*)alloc((size_t)TT * BB * 4608 * 4);
    unsigned short* hbf[2];
    hbf[0] = (unsigned short*)alloc((size_t)BB * 1152 * 2);
    hbf[1] = (unsigned short*)alloc((size_t)BB * 1152 * 2);
    float *hS[3], *cS[3];
    for (int l = 0; l < 3; ++l) {
        hS[l] = (float*)alloc((size_t)BB * Hs[l] * 4);
        cS[l] = (float*)alloc((size_t)BB * Hs[l] * 4);
    }

    // ---- weight conversion to padded bf16 ----
    for (int l = 0; l < 3; ++l) {
        convert_pad_bf16_kernel<<<2048, 256, 0, stream>>>(Wih[l], G4[l], Is[l],
                                                          WihB[l], Grp[l], IKp[l]);
        convert_pad_bf16_kernel<<<2048, 256, 0, stream>>>(Whh[l], G4[l], Hs[l],
                                                          WhhB[l], Wrp[l], HKp[l]);
    }
    convert_pad_bf16_kernel<<<4096, 256, 0, stream>>>(decW, VV, EE, decWB, 33280, 416);

    // ---- embedding ----
    embed_kernel<<<TT * BB, 128, 0, stream>>>(tokens, embW, xb[0], 416);

    // ---- zero activation buffers (pad columns must be 0) ----
    zero_u16_kernel<<<2048, 256, 0, stream>>>(xb[1], (long)TT * BB * 1152);
    zero_u16_kernel<<<2048, 256, 0, stream>>>(xb[2], (long)TT * BB * 1152);
    zero_u16_kernel<<<1024, 256, 0, stream>>>(xb[3], (long)TT * BB * 416);

    // ---- layers ----
    for (int l = 0; l < 3; ++l) {
        const int H = Hs[l], Kp = HKp[l];

        // xW = x * Wih^T + bih + bhh   (f32, stride 4608)
        dim3 gg(TT * BB / 32, (G4[l] + 63) / 64);
        wmma_gemm_bias_kernel<<<gg, 32, 0, stream>>>(xb[l], IKp[l], WihB[l], IKp[l],
                                                     bih[l], bhh[l], xWbuf, 4608,
                                                     G4[l], IKp[l] / 32, 0);

        // reset feedback buffers (stride changes between layers -> stale pads)
        zero_u16_kernel<<<256, 256, 0, stream>>>(hbf[0], (long)BB * Kp);
        zero_u16_kernel<<<256, 256, 0, stream>>>(hbf[1], (long)BB * Kp);
        init_state_kernel<<<(BB * H + 255) / 256, 256, 0, stream>>>(
            h0[l], c0[l], H, Kp, hbf[0], hS[l], cS[l]);

        int chunks = (H + 15) / 16;
        for (int t = 0; t < TT; ++t) {
            lstm_step_kernel<<<chunks, 128, 0, stream>>>(
                xWbuf, t, WhhB[l], hbf[t & 1], Kp, H, Kp / 32,
                hbf[(t + 1) & 1], cS[l], hS[l], xb[l + 1]);
        }
    }

    // ---- decoder: d_out[T*B][V] = x3 * decW^T + dec_b (non-temporal stores) ----
    dim3 gd(TT * BB / 32, (VV + 63) / 64);
    wmma_gemm_bias_kernel<<<gd, 32, 0, stream>>>(xb[3], 416, decWB, 416,
                                                 decb, nullptr, (float*)d_out,
                                                 (long)VV, VV, 416 / 32, 1);

    // ---- final hidden/cell states to output tail ----
    float* outp = (float*)d_out + (size_t)TT * BB * VV;
    for (int l = 0; l < 3; ++l) {
        size_t n = (size_t)BB * Hs[l];
        hipMemcpyAsync(outp, hS[l], n * 4, hipMemcpyDeviceToDevice, stream); outp += n;
        hipMemcpyAsync(outp, cS[l], n * 4, hipMemcpyDeviceToDevice, stream); outp += n;
    }
}
